// MultiHeadAttention_59253368815813
// MI455X (gfx1250) — matrix operand order repaired and checked by reading, not compiled, not run on hardware
//
#include <hip/hip_runtime.h>

typedef __bf16 v16bf __attribute__((ext_vector_type(16)));
typedef float  v8f   __attribute__((ext_vector_type(8)));
typedef unsigned short u16;
typedef unsigned int   u32;

union FragB16 { u32 w[8]; v16bf v; };

__device__ __forceinline__ u16 f2bf(float f) {
  u32 u = __builtin_bit_cast(u32, f);
  u32 r = u + 0x7FFFu + ((u >> 16) & 1u);
  return (u16)(r >> 16);
}

// A-matrix (16x32 bf16) K-start for VGPR i, lane-half h (contiguous: k = 16h + 2i)
__device__ __forceinline__ int kstartA(int i, int half) {
  return 16 * half + 2 * i;
}

__device__ __forceinline__ v8f wmma_bf16(const FragB16& a, const FragB16& b, v8f c) {
  return __builtin_amdgcn_wmma_f32_16x16x32_bf16(
      false, a.v, false, b.v, (short)0, c, false, false);
}

// ---- CDNA5 async global->LDS copy (ASYNCcnt path), per-lane 16B ----
__device__ __forceinline__ void async_b128(u32 lds_off, const u16* gaddr) {
  asm volatile("global_load_async_to_lds_b128 %0, %1, off"
               :: "v"(lds_off), "v"(gaddr)
               : "memory");
}
// partial wait: async loads retire in order; each stage = 4 ops/wave,
// so <=4 outstanding means the oldest stage has landed in LDS.
__device__ __forceinline__ void wait_async_le4() {
  asm volatile("s_wait_asynccnt 0x4" ::: "memory");
}
__device__ __forceinline__ void wait_async_0() {
  asm volatile("s_wait_asynccnt 0x0" ::: "memory");
}
// generic-pointer low 32 bits == wave-relative LDS offset (ISA 10.2 aperture rule)
__device__ __forceinline__ u32 lds_off_of(const void* p) {
  return (u32)(uintptr_t)p;
}

// ---------------- conversion kernels ----------------
__global__ void cvt_f32_bf16(const float* __restrict__ s, u16* __restrict__ d, int n) {
  int i = blockIdx.x * blockDim.x + threadIdx.x;
  if (i < n) d[i] = f2bf(s[i]);
}

__global__ void cvt_transpose_f32_bf16(const float* __restrict__ s, u16* __restrict__ d,
                                       int R, int C) {
  int i = blockIdx.x * blockDim.x + threadIdx.x;
  if (i < R * C) {
    int r = i / C, c = i % C;
    d[(size_t)c * R + r] = f2bf(s[(size_t)r * C + c]);
  }
}

// ---------------- bf16 WMMA GEMM, 3-stage async LDS pipeline ----------------
// C[M,N] = A[M,K] * Bt[N,K]^T * scale   (Bt is B stored column-major: Bt[n][k])
// MODE 0: fp32 out (Cf), MODE 1: bf16 out (Cb), MODE 2: bf16 transposed out Cb[n*M+m]
#define BM 128
#define BN 128
#define BK 32
#define NBUF 3
#define LPITCH 40   // ushorts; 80B row pitch: 16B-aligned, 20 dwords -> conflict-free

template <int MODE>
__global__ __launch_bounds__(256)
void gemm_bf16_wmma(const u16* __restrict__ A, const u16* __restrict__ Bt,
                    float* __restrict__ Cf, u16* __restrict__ Cb,
                    int M, int N, int K, float scale)
{
  struct SMTile { u16 A[BM][LPITCH]; u16 B[BN][LPITCH]; };
  __shared__ SMTile sm[NBUF];

  const int tid  = threadIdx.x;
  const int lane = tid & 31;
  const int wave = tid >> 5;
  const int l15  = lane & 15;
  const int half = lane >> 4;
  const int wm   = wave & 3;       // 4 waves tile M (32 rows each)
  const int wn   = wave >> 2;      // 2 waves tile N (64 cols each)
  const int m0   = blockIdx.y * BM;
  const int n0   = blockIdx.x * BN;

  v8f acc[2][4];
#pragma unroll
  for (int mt = 0; mt < 2; ++mt)
#pragma unroll
    for (int nt = 0; nt < 4; ++nt) acc[mt][nt] = (v8f)(0.0f);

  const int srow = tid >> 1;          // 0..127
  const int skh  = (tid & 1) * 16;    // 0 or 16
  const u16* gA = &A [(size_t)(m0 + srow) * K + skh];
  const u16* gB = &Bt[(size_t)(n0 + srow) * K + skh];

  auto issue_stage = [&](int buf, int k0) {   // 4 async ops per wave per stage
    async_b128(lds_off_of(&sm[buf].A[srow][skh]),     gA + k0);
    async_b128(lds_off_of(&sm[buf].A[srow][skh + 8]), gA + k0 + 8);
    async_b128(lds_off_of(&sm[buf].B[srow][skh]),     gB + k0);
    async_b128(lds_off_of(&sm[buf].B[srow][skh + 8]), gB + k0 + 8);
  };

  const int NK = K / BK;
  issue_stage(0, 0);
  if (NK > 1) issue_stage(1, BK);

  for (int kt = 0; kt < NK; ++kt) {
    if (kt + 1 < NK) wait_async_le4();   // retire stage kt, keep kt+1 in flight
    else             wait_async_0();
    __syncthreads();
    if (kt + 2 < NK) issue_stage((kt + 2) % NBUF, (kt + 2) * BK);

    const SMTile& t = sm[kt % NBUF];
    FragB16 a[2], b[4];
#pragma unroll
    for (int mt = 0; mt < 2; ++mt) {
      const int m = wm * 32 + mt * 16 + l15;
#pragma unroll
      for (int i = 0; i < 8; ++i)
        a[mt].w[i] = *reinterpret_cast<const u32*>(&t.A[m][kstartA(i, half)]);
    }
#pragma unroll
    for (int nt = 0; nt < 4; ++nt) {
      const int n = wn * 64 + nt * 16 + l15;
#pragma unroll
      for (int i = 0; i < 8; ++i)
        b[nt].w[i] = *reinterpret_cast<const u32*>(&t.B[n][16 * half + 2 * i]);
    }
#pragma unroll
    for (int mt = 0; mt < 2; ++mt)
#pragma unroll
      for (int nt = 0; nt < 4; ++nt)
        acc[mt][nt] = wmma_bf16(a[mt], b[nt], acc[mt][nt]);
  }

#pragma unroll
  for (int mt = 0; mt < 2; ++mt)
#pragma unroll
    for (int nt = 0; nt < 4; ++nt) {
      const int colg = n0 + wn * 64 + nt * 16 + l15;
#pragma unroll
      for (int r = 0; r < 8; ++r) {
        const int rowg = m0 + wm * 32 + mt * 16 + r + 8 * half;
        const float v = acc[mt][nt][r] * scale;
        if (MODE == 0)      Cf[(size_t)rowg * N + colg] = v;
        else if (MODE == 1) Cb[(size_t)rowg * N + colg] = f2bf(v);
        else                Cb[(size_t)colg * M + rowg] = f2bf(v);
      }
    }
}

// ---------------- flash attention (1 wave / 16-query tile / head / batch) ----------------
__global__ __launch_bounds__(32)
void attn_wmma(const u16* __restrict__ Qm, const u16* __restrict__ Km,
               const u16* __restrict__ VT, const float* __restrict__ mask,
               u16* __restrict__ Om)
{
  __shared__ u16 Pl[16][36];

  const int lane = threadIdx.x;
  const int l15  = lane & 15;
  const int half = lane >> 4;
  const int t0 = blockIdx.x * 16;
  const int h  = blockIdx.y;
  const int bi = blockIdx.z;
  const int T = 2048, D = 1024, DK = 64;
  const size_t rowbase = (size_t)bi * T;

  // Q A-fragments (16x64 over two k=32 chunks), resident for whole kernel
  FragB16 qa[2];
#pragma unroll
  for (int c = 0; c < 2; ++c)
#pragma unroll
    for (int i = 0; i < 8; ++i) {
      const int k = 32 * c + kstartA(i, half);
      qa[c].w[i] = *reinterpret_cast<const u32*>(
          &Qm[(rowbase + t0 + l15) * D + h * DK + k]);
    }

  v8f oacc[4];
  float mrun[8], lrun[8];
#pragma unroll
  for (int g = 0; g < 4; ++g) oacc[g] = (v8f)(0.0f);
#pragma unroll
  for (int r = 0; r < 8; ++r) { mrun[r] = -3.0e38f; lrun[r] = 0.0f; }

  for (int key0 = 0; key0 < T; key0 += 32) {
    // ---- S = Q K^T for 32 keys (two 16-key column groups) ----
    v8f s[2];
#pragma unroll
    for (int sg = 0; sg < 2; ++sg) {
      FragB16 kb0, kb1;
      const size_t krow = (rowbase + key0 + 16 * sg + l15) * D + h * DK;
#pragma unroll
      for (int i = 0; i < 8; ++i) {
        const int kk = 16 * half + 2 * i;
        kb0.w[i] = *reinterpret_cast<const u32*>(&Km[krow + kk]);
        kb1.w[i] = *reinterpret_cast<const u32*>(&Km[krow + 32 + kk]);
      }
      v8f z = (v8f)(0.0f);
      z = wmma_bf16(qa[0], kb0, z);
      z = wmma_bf16(qa[1], kb1, z);
      const float mv = mask[rowbase + key0 + 16 * sg + l15];
#pragma unroll
      for (int r = 0; r < 8; ++r) z[r] += mv;
      s[sg] = z;
    }
    // ---- online softmax (row reductions across the 16 lanes of each half) ----
#pragma unroll
    for (int r = 0; r < 8; ++r) {
      float tm = fmaxf(s[0][r], s[1][r]);
#pragma unroll
      for (int off = 1; off < 16; off <<= 1)
        tm = fmaxf(tm, __shfl_xor(tm, off, 32));
      const float nm   = fmaxf(mrun[r], tm);
      const float corr = __expf(mrun[r] - nm);
      mrun[r] = nm;
      const float p0 = __expf(s[0][r] - nm);
      const float p1 = __expf(s[1][r] - nm);
      s[0][r] = p0; s[1][r] = p1;
      float ls = p0 + p1;
#pragma unroll
      for (int off = 1; off < 16; off <<= 1)
        ls += __shfl_xor(ls, off, 32);
      lrun[r] = lrun[r] * corr + ls;
#pragma unroll
      for (int g = 0; g < 4; ++g) oacc[g][r] *= corr;
    }
    // ---- P (C-layout fp32) -> A-fragment bf16 via LDS ----
#pragma unroll
    for (int sg = 0; sg < 2; ++sg)
#pragma unroll
      for (int r = 0; r < 8; ++r)
        Pl[r + 8 * half][16 * sg + l15] = f2bf(s[sg][r]);
    __syncthreads();
    FragB16 pa;
#pragma unroll
    for (int i = 0; i < 8; ++i)
      pa.w[i] = *reinterpret_cast<const u32*>(&Pl[l15][kstartA(i, half)]);
    __syncthreads();
    // ---- O += P V  (V pre-transposed: VT[d][4096] -> packed key-pairs) ----
#pragma unroll
    for (int g = 0; g < 4; ++g) {
      FragB16 vb;
      const size_t vcol = (size_t)(h * DK + 16 * g + l15) * (2 * T) + rowbase + key0;
#pragma unroll
      for (int i = 0; i < 8; ++i)
        vb.w[i] = *reinterpret_cast<const u32*>(&VT[vcol + 16 * half + 2 * i]);
      oacc[g] = wmma_bf16(pa, vb, oacc[g]);
    }
  }

  // store in the reference's faithful scrambled reshape:
  // row = bi*2048 + h*128 + t0/16, col = (t%16)*64 + dc
  const size_t orow = (rowbase + h * 128 + (t0 >> 4)) * D;
#pragma unroll
  for (int g = 0; g < 4; ++g)
#pragma unroll
    for (int r = 0; r < 8; ++r) {
      const float v = oacc[g][r] / lrun[r];
      Om[orow + (size_t)(r + 8 * half) * DK + 16 * g + l15] = f2bf(v);
    }
}

// ---------------- host launcher ----------------
extern "C" void kernel_launch(void* const* d_in, const int* in_sizes, int n_in,
                              void* d_out, int out_size, void* d_ws, size_t ws_size,
                              hipStream_t stream) {
  const float* x    = (const float*)d_in[0];
  const float* mask = (const float*)d_in[1];
  const float* Wq   = (const float*)d_in[2];
  const float* Wk   = (const float*)d_in[3];
  const float* Wv   = (const float*)d_in[4];
  const float* Wo   = (const float*)d_in[5];
  float* out = (float*)d_out;

  const int B = 2, T = 2048, D = 1024;
  const int Mrows = B * T;                 // 4096

  u16* ws  = (u16*)d_ws;
  u16* xb  = ws;                           // [4096][1024] bf16
  u16* wqb = xb  + (size_t)Mrows * D;      // Wq^T as Bt[n][k]
  u16* wkb = wqb + (size_t)D * D;          // Wk^T as Bt[n][k]
  u16* wvb = wkb + (size_t)D * D;          // Wv^T as Bt[n][k]
  u16* wob = wvb + (size_t)D * D;          // Wo (== (Wo^T)^T) as Bt[j][k]
  u16* Qb  = wob + (size_t)D * D;          // [4096][1024] (scaled)
  u16* Kb  = Qb  + (size_t)Mrows * D;      // [4096][1024]
  u16* VTb = Kb  + (size_t)Mrows * D;      // [1024][4096] transposed
  u16* Ob  = VTb + (size_t)Mrows * D;      // [4096][1024] scrambled reshape

  const int nx = Mrows * D;
  const int nw = D * D;
  cvt_f32_bf16<<<(nx + 255) / 256, 256, 0, stream>>>(x, xb, nx);
  cvt_transpose_f32_bf16<<<(nw + 255) / 256, 256, 0, stream>>>(Wq, wqb, D, D);
  cvt_transpose_f32_bf16<<<(nw + 255) / 256, 256, 0, stream>>>(Wk, wkb, D, D);
  cvt_transpose_f32_bf16<<<(nw + 255) / 256, 256, 0, stream>>>(Wv, wvb, D, D);
  cvt_f32_bf16<<<(nw + 255) / 256, 256, 0, stream>>>(Wo, wob, nw);

  dim3 gg(D / BN, Mrows / BM);             // (8, 32)
  const float qscale = 0.125f;             // 1/sqrt(dk), dk = 64
  gemm_bf16_wmma<1><<<gg, 256, 0, stream>>>(xb, wqb, nullptr, Qb,  Mrows, D, D, qscale);
  gemm_bf16_wmma<1><<<gg, 256, 0, stream>>>(xb, wkb, nullptr, Kb,  Mrows, D, D, 1.0f);
  gemm_bf16_wmma<2><<<gg, 256, 0, stream>>>(xb, wvb, nullptr, VTb, Mrows, D, D, 1.0f);

  attn_wmma<<<dim3(T / 16, 16, B), 32, 0, stream>>>(Qb, Kb, VTb, mask, Ob);

  gemm_bf16_wmma<0><<<gg, 256, 0, stream>>>(Ob, wob, out, nullptr, Mrows, D, D, 1.0f);
}
